// TransducerJoint_84567906058998
// MI455X (gfx1250) — matrix operand order repaired and checked by reading, not compiled, not run on hardware
//
#include <hip/hip_runtime.h>
#include <hip/hip_bf16.h>
#include <math.h>

// ---- problem constants (from reference setup_inputs) ----
#define B_   4
#define T_   200
#define U_   100
#define E_   512
#define P_   512
#define D_   512
#define V_   2048
#define ROWS (B_ * T_ * U_)   // 80000 flattened (b,t,u) rows

typedef __attribute__((ext_vector_type(16))) __bf16 v16bf;
typedef __attribute__((ext_vector_type(8)))  float  v8f;

__device__ __forceinline__ float fast_tanhf(float x) {
#if __has_builtin(__builtin_amdgcn_tanhf)
  return __builtin_amdgcn_tanhf(x);      // gfx1250 v_tanh_f32 (TRANS unit)
#elif __has_builtin(__builtin_amdgcn_tanh_f32)
  return __builtin_amdgcn_tanh_f32(x);
#else
  return tanhf(x);
#endif
}

// round-to-nearest-even fp32 -> bf16, packed pair into one dword
__device__ __forceinline__ unsigned int pack_bf16x2(float a, float b) {
  unsigned int ua = __float_as_uint(a);
  unsigned int ub = __float_as_uint(b);
  ua = (ua + 0x7FFFu + ((ua >> 16) & 1u)) >> 16;
  ub = (ub + 0x7FFFu + ((ub >> 16) & 1u)) >> 16;
  return (ua & 0xFFFFu) | (ub << 16);
}

// ---------------------------------------------------------------------------
// Kernel 1/2: y[r,d] = dot(x[r,:], w[d,:]) + bias[d]   (small projections)
// ---------------------------------------------------------------------------
__global__ void proj_kernel(const float* __restrict__ x,   // [rows, K]
                            const float* __restrict__ w,   // [D, K]
                            const float* __restrict__ bias,// [D]
                            float* __restrict__ y,         // [rows, D]
                            int K) {
  int idx = blockIdx.x * blockDim.x + threadIdx.x;
  int r = idx / D_;
  int d = idx - r * D_;
  const float4* xp = (const float4*)(x + (size_t)r * K);
  const float4* wp = (const float4*)(w + (size_t)d * K);
  float acc = 0.f;
  int n4 = K >> 2;
  for (int i = 0; i < n4; ++i) {
    float4 a = xp[i], b = wp[i];
    acc += a.x * b.x + a.y * b.y + a.z * b.z + a.w * b.w;
  }
  y[idx] = acc + bias[d];
}

// ---------------------------------------------------------------------------
// Kernel 3: pre-pack out_w [V,D] fp32 -> bf16 WMMA B-fragments.
// Fragment (nt, ks) covers cols v = 16*nt..+15, k = 32*ks..+31.
// Contiguous-K B-layout used here (16-bit, 32x16): lane<16 -> N=lane, K rel 0..15 in 8 dwords;
// lane>=16 -> N=lane-16, K rel 16..31. Each lane: 32 bytes (16 bf16).
// ---------------------------------------------------------------------------
__global__ void packw_kernel(const float* __restrict__ w,  // [V, D]
                             unsigned int* __restrict__ wp) {
  int idx = blockIdx.x * blockDim.x + threadIdx.x;   // = (nt*16 + ks)*32 + lane
  int lane = idx & 31;
  int ks   = (idx >> 5) & 15;
  int nt   = idx >> 9;
  int v     = nt * 16 + (lane & 15);
  int kbase = ks * 32 + ((lane < 16) ? 0 : 16);
  const float* src = w + (size_t)v * D_ + kbase;
  unsigned int d[8];
#pragma unroll
  for (int j = 0; j < 8; ++j) d[j] = pack_bf16x2(src[2 * j], src[2 * j + 1]);
  unsigned int* dst = wp + (size_t)idx * 8;
  ((int4*)dst)[0] = make_int4(d[0], d[1], d[2], d[3]);
  ((int4*)dst)[1] = make_int4(d[4], d[5], d[6], d[7]);
}

// ---------------------------------------------------------------------------
// Kernel 4: fused join + vocab GEMM.
// WG = 256 threads (8 waves). Output tile: M=32 rows x N=256 cols.
// Phase 1: compute A = tanh(e+p) in bf16, stored to LDS pre-swizzled into
//          WMMA A-fragment layout (contiguous k = 16*(lane/16) + s, same as B pack).
// Phase 2: each wave: 2 M-subtiles x 2 N-subtiles, K-loop of 16 steps,
//          4x v_wmma_f32_16x16x32_bf16 per step.
// ---------------------------------------------------------------------------
__global__ __launch_bounds__(256)
void joint_kernel(const float* __restrict__ e,          // [B*T, D]
                  const float* __restrict__ p,          // [B*U, D]
                  const unsigned int* __restrict__ wpack,
                  const float* __restrict__ out_b,      // [V]
                  float* __restrict__ out) {            // [ROWS, V]
  __shared__ unsigned int lds[32 * 32 * 8];             // 32 KB: 32 frags x 32 lanes x 8 dw
  const int tid     = threadIdx.x;
  const int rowBase = blockIdx.x * 32;
  const int vBase   = blockIdx.y * 256;

  // ---- Phase 1: build swizzled A tile (32 rows x 512 K, bf16) ----
#pragma unroll
  for (int it = 0; it < 4; ++it) {
    int c    = tid + 256 * it;      // chunk = frag*32 + lane
    int lane = c & 31;
    int frag = c >> 5;              // mt*16 + ks
    int ks   = frag & 15;
    int mt   = frag >> 4;
    int row  = rowBase + mt * 16 + (lane & 15);
    int b    = row / (T_ * U_);
    int rem  = row - b * (T_ * U_);
    int t    = rem / U_;
    int u    = rem - t * U_;
    const float* ep = e + (size_t)(b * T_ + t) * D_;
    const float* pp = p + (size_t)(b * U_ + u) * D_;
    int base0 = (lane < 16) ? 0 : 16;         // A layout K split within k-step
    int kA = ks * 32 + base0;                 // 8 contiguous k's
    int kB = kA + 8;                          // next 8 contiguous k's
    float ha[8], hb[8];
#pragma unroll
    for (int j = 0; j < 8; ++j) ha[j] = fast_tanhf(ep[kA + j] + pp[kA + j]);
#pragma unroll
    for (int j = 0; j < 8; ++j) hb[j] = fast_tanhf(ep[kB + j] + pp[kB + j]);
    unsigned int d[8];
#pragma unroll
    for (int j = 0; j < 4; ++j) d[j]     = pack_bf16x2(ha[2 * j], ha[2 * j + 1]);
#pragma unroll
    for (int j = 0; j < 4; ++j) d[4 + j] = pack_bf16x2(hb[2 * j], hb[2 * j + 1]);
    unsigned int* dst = &lds[c * 8];
    ((int4*)dst)[0] = make_int4(d[0], d[1], d[2], d[3]);
    ((int4*)dst)[1] = make_int4(d[4], d[5], d[6], d[7]);
  }
  __syncthreads();

  // ---- Phase 2: WMMA main loop ----
  const int lane = tid & 31;
  const int wave = tid >> 5;
  const int nt0  = (vBase >> 4) + wave * 2;             // global 16-col tile idx
  const unsigned int* wp0 = wpack + (size_t)(nt0 * 16) * 32 * 8;
  const unsigned int* wp1 = wpack + (size_t)((nt0 + 1) * 16) * 32 * 8;

  union Frag { int4 i[2]; v16bf v; };
  v8f c00 = {}, c01 = {}, c10 = {}, c11 = {};

  for (int ks = 0; ks < 16; ++ks) {
    Frag a0, a1, b0, b1;
    const unsigned int* ap0 = &lds[(ks * 32 + lane) * 8];
    const unsigned int* ap1 = &lds[((16 + ks) * 32 + lane) * 8];
    a0.i[0] = ((const int4*)ap0)[0];  a0.i[1] = ((const int4*)ap0)[1];
    a1.i[0] = ((const int4*)ap1)[0];  a1.i[1] = ((const int4*)ap1)[1];
    const unsigned int* bp0 = wp0 + (size_t)(ks * 32 + lane) * 8;
    const unsigned int* bp1 = wp1 + (size_t)(ks * 32 + lane) * 8;
    b0.i[0] = ((const int4*)bp0)[0];  b0.i[1] = ((const int4*)bp0)[1];
    b1.i[0] = ((const int4*)bp1)[0];  b1.i[1] = ((const int4*)bp1)[1];
    if (ks + 1 < 16) {   // hide next B-fragment latency (global_prefetch_b8)
      __builtin_prefetch(wp0 + (size_t)((ks + 1) * 32 + lane) * 8, 0, 1);
      __builtin_prefetch(wp1 + (size_t)((ks + 1) * 32 + lane) * 8, 0, 1);
    }
    c00 = __builtin_amdgcn_wmma_f32_16x16x32_bf16(false, a0.v, false, b0.v, (short)0, c00, false, false);
    c01 = __builtin_amdgcn_wmma_f32_16x16x32_bf16(false, a0.v, false, b1.v, (short)0, c01, false, false);
    c10 = __builtin_amdgcn_wmma_f32_16x16x32_bf16(false, a1.v, false, b0.v, (short)0, c10, false, false);
    c11 = __builtin_amdgcn_wmma_f32_16x16x32_bf16(false, a1.v, false, b1.v, (short)0, c11, false, false);
  }

  // ---- Epilogue: C layout -> VGPR i holds M = i + 8*(lane>=16), N = lane%16 ----
  const int v0 = nt0 * 16 + (lane & 15);
  const int v1 = v0 + 16;
  const float bias0 = out_b[v0];
  const float bias1 = out_b[v1];
  const int mBase = rowBase + ((lane >> 4) << 3);
#pragma unroll
  for (int i = 0; i < 8; ++i) {
    size_t r0 = (size_t)(mBase + i) * V_;        // M-subtile 0
    size_t r1 = (size_t)(mBase + 16 + i) * V_;   // M-subtile 1
    out[r0 + v0] = c00[i] + bias0;
    out[r0 + v1] = c01[i] + bias1;
    out[r1 + v0] = c10[i] + bias0;
    out[r1 + v1] = c11[i] + bias1;
  }
}

// ---------------------------------------------------------------------------
extern "C" void kernel_launch(void* const* d_in, const int* in_sizes, int n_in,
                              void* d_out, int out_size, void* d_ws, size_t ws_size,
                              hipStream_t stream) {
  const float* enc_out  = (const float*)d_in[0];  // [B,T,E]
  const float* pred_out = (const float*)d_in[1];  // [B,U,P]
  const float* enc_w    = (const float*)d_in[2];  // [D,E]
  const float* enc_b    = (const float*)d_in[3];  // [D]
  const float* pred_w   = (const float*)d_in[4];  // [D,P]
  const float* pred_b   = (const float*)d_in[5];  // [D]
  const float* out_w    = (const float*)d_in[6];  // [V,D]
  const float* out_b    = (const float*)d_in[7];  // [V]
  float* out = (float*)d_out;

  // workspace layout (all 16B aligned)
  char* ws = (char*)d_ws;
  float* e = (float*)ws;                                     // B*T*D fp32 (1.6 MB)
  float* p = (float*)(ws + (size_t)B_ * T_ * D_ * 4);        // B*U*D fp32 (0.8 MB)
  unsigned int* wpack =
      (unsigned int*)(ws + (size_t)B_ * T_ * D_ * 4 + (size_t)B_ * U_ * D_ * 4); // 2 MB

  proj_kernel<<<(B_ * T_ * D_) / 256, 256, 0, stream>>>(enc_out, enc_w, enc_b, e, E_);
  proj_kernel<<<(B_ * U_ * D_) / 256, 256, 0, stream>>>(pred_out, pred_w, pred_b, p, P_);
  packw_kernel<<<(V_ / 16) * (D_ / 32) * 32 / 256, 256, 0, stream>>>(out_w, wpack);

  dim3 grid(ROWS / 32, V_ / 256);   // 2500 x 8 workgroups
  joint_kernel<<<grid, 256, 0, stream>>>(e, p, wpack, out_b, out);
}
